// HTHGNLayer_53626961658092
// MI455X (gfx1250) — matrix operand order repaired and checked by reading, not compiled, not run on hardware
//
#include <hip/hip_runtime.h>
#include <hip/hip_bf16.h>
#include <math.h>

typedef __bf16 bf16;
typedef __attribute__((ext_vector_type(16))) __bf16 v16bf;
typedef __attribute__((ext_vector_type(8)))  float  v8f;

#define T_  6
#define N_  50000
#define NH_ 128
#define H_  4
#define D_  32
#define R_  2
#define E_  800000

// ---------------------------------------------------------------------------
// float atomic max via int-ordered trick (resolves at L2)
// ---------------------------------------------------------------------------
__device__ __forceinline__ void atomic_max_f32(float* addr, float val) {
  if (val >= 0.f) atomicMax((int*)addr, __float_as_int(val));
  else            atomicMin((unsigned int*)addr, (unsigned int)__float_as_int(val));
}

// ---------------------------------------------------------------------------
// weight f32 -> bf16 conversion, TRANSPOSED: dst[col][k] = src[k][col].
// K-contiguous columns let the GEMM load B fragments as 2x ds_load_b128.
// ---------------------------------------------------------------------------
__global__ void k_f32_to_bf16_T(const float* __restrict__ src, bf16* __restrict__ dst) {
  int i = blockIdx.x * 256 + threadIdx.x;          // i = k*128 + c
  if (i < NH_ * NH_) {
    const int k = i >> 7, c = i & 127;
    dst[c * NH_ + k] = (bf16)src[i];
  }
}

// ---------------------------------------------------------------------------
// Generic GEMM: C[M,128] = A[M,128] @ W[128,128], bf16 WMMA, f32 accumulate.
// One wave computes a 16-row x 128-col strip: 8 accumulators, 4 K-steps.
// W arrives pre-transposed bf16 [col][k]; staged to LDS as-is.
// Layouts (C per CDNA5 ISA 7.12.2; A/B use this file's contiguous k order):
//   A 16x32 bf16 : lanes 0-15 row M=lr, elems = K{half*16 .. half*16+15}
//   B 32x16 bf16 : lanes hold col N=lr, elems = K{half*16 .. half*16+15}
//   C 16x16 f32  : lane -> N = lr, vgpr j -> M = j + half*8
// epi: 0=none  1=+bias  2=relu(+bias)  3=+bias+pe[row/N_]  4=semantic row-dot
// ---------------------------------------------------------------------------
__global__ __launch_bounds__(256) void k_gemm128(
    const float* __restrict__ A, const bf16* __restrict__ Wt,
    float* __restrict__ C, int M, int epi,
    const float* __restrict__ bias, const float* __restrict__ pe,
    const float* __restrict__ w2, float* __restrict__ wsum)
{
  __shared__ __align__(32) bf16 Ws[NH_ * NH_];     // transposed: [col][k]
  {
    const uint4* s4 = (const uint4*)Wt;
    uint4*       d4 = (uint4*)Ws;
    for (int i = threadIdx.x; i < (NH_ * NH_) / 8; i += 256) d4[i] = s4[i];
  }
  __syncthreads();

  const int wave = threadIdx.x >> 5;
  const int lane = threadIdx.x & 31;
  const int half = lane >> 4;
  const int lr   = lane & 15;
  const int m0   = (blockIdx.x * 8 + wave) * 16;
  if (m0 >= M) return;

  v8f acc[8];
#pragma unroll
  for (int n = 0; n < 8; ++n)
#pragma unroll
    for (int j = 0; j < 8; ++j) acc[n][j] = 0.f;

  const float* Arow = A + (size_t)(m0 + lr) * NH_;

#pragma unroll
  for (int kc = 0; kc < 4; ++kc) {
    v16bf af;
    const int ka = kc * 32 + half * 16;
#pragma unroll
    for (int e = 0; e < 8; ++e) af[e]     = (bf16)Arow[ka + e];
#pragma unroll
    for (int e = 0; e < 8; ++e) af[8 + e] = (bf16)Arow[ka + 8 + e];

    const int kb = kc * 32 + half * 16;
#pragma unroll
    for (int n = 0; n < 8; ++n) {
      const int col = n * 16 + lr;
      // 16 contiguous bf16 (32B, aligned) -> 2x ds_load_b128
      const v16bf bfr = *(const v16bf*)(Ws + (size_t)col * NH_ + kb);
      acc[n] = __builtin_amdgcn_wmma_f32_16x16x32_bf16(
          false, af, false, bfr, (short)0, acc[n], false, false);
    }
  }

  if (epi == 4) {
    // semantic attention: w_row = tanh(row + b1) . w2 ; accumulate mean per r
    float part[8];
#pragma unroll
    for (int j = 0; j < 8; ++j) part[j] = 0.f;
#pragma unroll
    for (int n = 0; n < 8; ++n) {
      const int col = n * 16 + lr;
      const float b1 = bias[col], w2c = w2[col];
#pragma unroll
      for (int j = 0; j < 8; ++j) part[j] += tanhf(acc[n][j] + b1) * w2c;
    }
#pragma unroll
    for (int off = 1; off < 16; off <<= 1)
#pragma unroll
      for (int j = 0; j < 8; ++j) part[j] += __shfl_xor(part[j], off, 32);
    if (lr == 0) {
      float s0 = 0.f, s1 = 0.f;
#pragma unroll
      for (int j = 0; j < 8; ++j) {
        const int row = m0 + j + half * 8;   // row = n*2 + r  -> r = row & 1
        if (row & 1) s1 += part[j]; else s0 += part[j];
      }
      atomicAdd(&wsum[0], s0);
      atomicAdd(&wsum[1], s1);
    }
    return;
  }

#pragma unroll
  for (int n = 0; n < 8; ++n) {
    const int col = n * 16 + lr;
    const float bv = (epi >= 1) ? bias[col] : 0.f;
#pragma unroll
    for (int j = 0; j < 8; ++j) {
      const int row = m0 + j + half * 8;
      float cv = acc[n][j] + bv;
      if (epi == 3) cv += pe[(row / N_) * NH_ + col];
      if (epi == 2) cv = fmaxf(cv, 0.f);
      C[(size_t)row * NH_ + col] = cv;
    }
  }
}

// ---------------------------------------------------------------------------
// GAT edge phase
// ---------------------------------------------------------------------------
__global__ void k_init_h(float* __restrict__ h, const float* __restrict__ gat_bias,
                         float* __restrict__ wsum) {
  int i = blockIdx.x * 256 + threadIdx.x;
  if (i == 0) { wsum[0] = 0.f; wsum[1] = 0.f; }
  if (i < N_ * R_ * NH_) h[i] = gat_bias[i & (R_ * NH_ - 1)];
}

__global__ void k_init_maxden(float* __restrict__ maxb, float* __restrict__ den) {
  int i = blockIdx.x * 256 + threadIdx.x;
  if (i < N_ * H_) { maxb[i] = __int_as_float(0xFF800000); den[i] = 0.f; }
}

__global__ __launch_bounds__(256) void k_edge_score(
    const float* __restrict__ fs, const float* __restrict__ fd,
    const int* __restrict__ src, const int* __restrict__ dst,
    const float* __restrict__ attn, float* __restrict__ sc, float* __restrict__ maxb)
{
  int e = blockIdx.x * 256 + threadIdx.x;
  if (e >= E_) return;
  const int s = src[e], d = dst[e];
  const float4* fs4 = (const float4*)(fs + (size_t)s * NH_);
  const float4* fd4 = (const float4*)(fd + (size_t)d * NH_);
  const float4* at4 = (const float4*)attn;
#pragma unroll
  for (int h = 0; h < H_; ++h) {
    float acc = 0.f;
#pragma unroll
    for (int q = 0; q < 8; ++q) {
      float4 a = fs4[h * 8 + q], b = fd4[h * 8 + q], w = at4[h * 8 + q];
      float x;
      x = a.x + b.x; x = x > 0.f ? x : 0.2f * x; acc += x * w.x;
      x = a.y + b.y; x = x > 0.f ? x : 0.2f * x; acc += x * w.y;
      x = a.z + b.z; x = x > 0.f ? x : 0.2f * x; acc += x * w.z;
      x = a.w + b.w; x = x > 0.f ? x : 0.2f * x; acc += x * w.w;
    }
    sc[(size_t)e * H_ + h] = acc;
    atomic_max_f32(&maxb[d * H_ + h], acc);
  }
}

__global__ void k_edge_exp(const int* __restrict__ dst, float* __restrict__ sc,
                           const float* __restrict__ maxb, float* __restrict__ den)
{
  int e = blockIdx.x * 256 + threadIdx.x;
  if (e >= E_) return;
  const int d = dst[e];
#pragma unroll
  for (int h = 0; h < H_; ++h) {
    float ex = __expf(sc[(size_t)e * H_ + h] - maxb[d * H_ + h]);
    sc[(size_t)e * H_ + h] = ex;
    atomicAdd(&den[d * H_ + h], ex);
  }
}

__global__ __launch_bounds__(256) void k_edge_agg(
    const float* __restrict__ fs, const int* __restrict__ src, const int* __restrict__ dst,
    const float* __restrict__ sc, const float* __restrict__ den,
    float* __restrict__ hout, int r)
{
  int e = blockIdx.x * 256 + threadIdx.x;
  if (e >= E_) return;
  const int s = src[e], d = dst[e];
  const float* fr = fs + (size_t)s * NH_;
  float* out = hout + ((size_t)d * R_ + r) * NH_;
#pragma unroll
  for (int h = 0; h < H_; ++h) {
    const float a = sc[(size_t)e * H_ + h] / den[d * H_ + h];
#pragma unroll
    for (int dd = 0; dd < D_; ++dd)
      atomicAdd(&out[h * D_ + dd], a * fr[h * D_ + dd]);
  }
}

// ---------------------------------------------------------------------------
// semantic beta + inter build
// ---------------------------------------------------------------------------
__global__ void k_sem_beta(const float* __restrict__ wsum, float* __restrict__ beta) {
  const float invN = 1.f / (float)N_;
  float w0 = wsum[0] * invN, w1 = wsum[1] * invN;
  float m = fmaxf(w0, w1);
  float e0 = __expf(w0 - m), e1 = __expf(w1 - m);
  float inv = 1.f / (e0 + e1);
  beta[0] = e0 * inv; beta[1] = e1 * inv;
}

__global__ void k_inter_build(const float* __restrict__ h, const float* __restrict__ beta,
                              const float* __restrict__ h_bias, float* __restrict__ inter_t)
{
  int i = blockIdx.x * 256 + threadIdx.x;
  if (i >= N_ * NH_) return;
  const int n = i >> 7, c = i & 127;
  const float hv = fmaxf(beta[0] * h[(size_t)n * 256 + c] +
                         beta[1] * h[(size_t)n * 256 + 128 + c], 0.f);
  inter_t[i] = fmaxf(hv + h_bias[c], 0.f);
}

// ---------------------------------------------------------------------------
// positional encoding [T,128]
// ---------------------------------------------------------------------------
__global__ void k_pe(float* __restrict__ pe) {
  int i = blockIdx.x * 256 + threadIdx.x;
  if (i >= T_ * NH_) return;
  const int t = i >> 7, c = i & 127;
  const float div = __expf(-(float)(c & ~1) * logf(100000.f) / (float)NH_);
  const float pos = (float)(t + 1);
  pe[i] = (c & 1) ? cosf(pos * div) : sinf(pos * div);
}

// ---------------------------------------------------------------------------
// temporal attention: one wave per node, 4 columns per lane, T=6
// (faithful: no 1/sqrt(d) scaling)
// ---------------------------------------------------------------------------
__global__ __launch_bounds__(256) void k_temp_attn(
    const float* __restrict__ Q, const float* __restrict__ K,
    const float* __restrict__ V, float* __restrict__ CTX)
{
  const int wave = threadIdx.x >> 5, lane = threadIdx.x & 31;
  const int n = blockIdx.x * 8 + wave;
  if (n >= N_) return;

  float q[T_][4], k[T_][4], v[T_][4];
#pragma unroll
  for (int t = 0; t < T_; ++t) {
    const size_t base = ((size_t)t * N_ + n) * NH_ + lane;
#pragma unroll
    for (int j = 0; j < 4; ++j) {
      q[t][j] = Q[base + 32 * j];
      k[t][j] = K[base + 32 * j];
      v[t][j] = V[base + 32 * j];
    }
  }
  float S[T_][T_];
#pragma unroll
  for (int t = 0; t < T_; ++t)
#pragma unroll
    for (int s = 0; s < T_; ++s) {
      float d = 0.f;
#pragma unroll
      for (int j = 0; j < 4; ++j) d += q[t][j] * k[s][j];
#pragma unroll
      for (int off = 16; off > 0; off >>= 1) d += __shfl_xor(d, off, 32);
      S[t][s] = d;
    }
#pragma unroll
  for (int t = 0; t < T_; ++t) {
    float mx = S[t][0];
#pragma unroll
    for (int s = 1; s < T_; ++s) mx = fmaxf(mx, S[t][s]);
    float dn = 0.f;
#pragma unroll
    for (int s = 0; s < T_; ++s) { float e = __expf(S[t][s] - mx); S[t][s] = e; dn += e; }
    const float inv = 1.f / dn;
#pragma unroll
    for (int s = 0; s < T_; ++s) S[t][s] *= inv;
  }
#pragma unroll
  for (int t = 0; t < T_; ++t) {
    const size_t base = ((size_t)t * N_ + n) * NH_ + lane;
#pragma unroll
    for (int j = 0; j < 4; ++j) {
      float c = 0.f;
#pragma unroll
      for (int s = 0; s < T_; ++s) c += S[t][s] * v[s][j];
      CTX[base + 32 * j] = c;
    }
  }
}

// ---------------------------------------------------------------------------
// gated residual + LayerNorm: one wave per row
// ---------------------------------------------------------------------------
__global__ __launch_bounds__(256) void k_final_ln(
    const float* __restrict__ HO, const float* __restrict__ RES,
    const float* __restrict__ alpha_p, const float* __restrict__ gamma,
    const float* __restrict__ betaLN, float* __restrict__ OUT)
{
  const int wave = threadIdx.x >> 5, lane = threadIdx.x & 31;
  const int row = blockIdx.x * 8 + wave;
  if (row >= T_ * N_) return;
  const float a = 1.f / (1.f + __expf(-alpha_p[0]));
  const size_t base = (size_t)row * NH_ + lane;

  float xv[4];
  float s = 0.f;
#pragma unroll
  for (int j = 0; j < 4; ++j) {
    xv[j] = HO[base + 32 * j] * a + RES[base + 32 * j] * (1.f - a);
    s += xv[j];
  }
#pragma unroll
  for (int off = 16; off > 0; off >>= 1) s += __shfl_xor(s, off, 32);
  const float mean = s * (1.f / NH_);
  float vs = 0.f;
#pragma unroll
  for (int j = 0; j < 4; ++j) { const float d = xv[j] - mean; vs += d * d; }
#pragma unroll
  for (int off = 16; off > 0; off >>= 1) vs += __shfl_xor(vs, off, 32);
  const float rstd = rsqrtf(vs * (1.f / NH_) + 1e-5f);
#pragma unroll
  for (int j = 0; j < 4; ++j) {
    const int c = lane + 32 * j;
    OUT[base + 32 * j] = (xv[j] - mean) * rstd * gamma[c] + betaLN[c];
  }
}

// ---------------------------------------------------------------------------
// launcher
// ---------------------------------------------------------------------------
extern "C" void kernel_launch(void* const* d_in, const int* in_sizes, int n_in,
                              void* d_out, int out_size, void* d_ws, size_t ws_size,
                              hipStream_t stream)
{
  const float* x        = (const float*)d_in[0];
  const int*   src      = (const int*)d_in[1];
  const int*   dst      = (const int*)d_in[2];
  const float* gat_wsrc = (const float*)d_in[3];
  const float* gat_wdst = (const float*)d_in[4];
  const float* gat_attn = (const float*)d_in[5];
  const float* gat_bias = (const float*)d_in[6];
  const float* h_bias   = (const float*)d_in[7];
  const float* sem_w1   = (const float*)d_in[8];
  const float* sem_b1   = (const float*)d_in[9];
  const float* sem_w2   = (const float*)d_in[10];
  const float* proj_w   = (const float*)d_in[11];
  const float* proj_b   = (const float*)d_in[12];
  const float* qw       = (const float*)d_in[13];
  const float* kw       = (const float*)d_in[14];
  const float* vw       = (const float*)d_in[15];
  const float* fc_w     = (const float*)d_in[16];
  const float* fc_b     = (const float*)d_in[17];
  const float* res_w    = (const float*)d_in[18];
  const float* res_b    = (const float*)d_in[19];
  const float* res_al   = (const float*)d_in[20];
  const float* ln_gamma = (const float*)d_in[21];
  const float* ln_beta  = (const float*)d_in[22];

  char* ws = (char*)d_ws;
  const size_t WMAT = (size_t)NH_ * NH_;           // 16384 elems
  bf16*  WBF  = (bf16*)ws;                         // 11 * 16384 bf16 (transposed)
  float* PE   = (float*)(ws + 360448);
  float* WSUM = (float*)(ws + 364544);
  float* BETA = (float*)(ws + 364608);
  const size_t BUFSZ = (size_t)T_ * N_ * NH_ * sizeof(float);
  float* BUF1 = (float*)(ws + (1 << 20));          // INTER -> Q -> HO
  float* BUF2 = (float*)(ws + (1 << 20) + BUFSZ);  // phaseA scratch -> HH -> CTX
  float* BUF3 = (float*)(ws + (1 << 20) + 2 * BUFSZ); // K -> RES
  float* VOUT = (float*)d_out;                     // V buffer, overwritten at end

  // phase-A scratch overlaid inside BUF2 (dead before HH is written)
  float* FS   = BUF2;
  float* FD   = BUF2 + (size_t)N_ * NH_;
  float* SC   = BUF2 + (size_t)2 * N_ * NH_;
  float* HB   = BUF2 + (size_t)2 * N_ * NH_ + (size_t)E_ * H_;
  float* MAXB = HB + (size_t)N_ * R_ * NH_;
  float* DEN  = MAXB + (size_t)N_ * H_;

  // ---- convert all 128x128 weights to transposed bf16 ----
  const float* wsrcs[11] = { gat_wsrc, gat_wsrc + WMAT, gat_wdst, gat_wdst + WMAT,
                             sem_w1, proj_w, qw, kw, vw, fc_w, res_w };
  for (int i = 0; i < 11; ++i)
    k_f32_to_bf16_T<<<64, 256, 0, stream>>>(wsrcs[i], WBF + (size_t)i * WMAT);
  bf16 *B_wsrc0 = WBF, *B_wsrc1 = WBF + WMAT, *B_wdst0 = WBF + 2 * WMAT,
       *B_wdst1 = WBF + 3 * WMAT, *B_sem = WBF + 4 * WMAT, *B_proj = WBF + 5 * WMAT,
       *B_q = WBF + 6 * WMAT, *B_k = WBF + 7 * WMAT, *B_v = WBF + 8 * WMAT,
       *B_fc = WBF + 9 * WMAT, *B_res = WBF + 10 * WMAT;

  k_pe<<<3, 256, 0, stream>>>(PE);

  auto gemm = [&](const float* A, const bf16* W, float* C, int M, int epi,
                  const float* bias, const float* pe, const float* w2, float* wsum) {
    const int blocks = (M / 16 + 7) / 8;
    k_gemm128<<<blocks, 256, 0, stream>>>(A, W, C, M, epi, bias, pe, w2, wsum);
  };

  const int eblk = (E_ + 255) / 256;

  // ---- phase A: per-timestep GAT + semantic aggregation ----
  for (int t = 0; t < T_; ++t) {
    const float* xt = x + (size_t)t * N_ * NH_;
    k_init_h<<<(N_ * R_ * NH_ + 255) / 256, 256, 0, stream>>>(HB, gat_bias, WSUM);
    for (int r = 0; r < R_; ++r) {
      gemm(xt, r ? B_wsrc1 : B_wsrc0, FS, N_, 0, nullptr, nullptr, nullptr, nullptr);
      gemm(xt, r ? B_wdst1 : B_wdst0, FD, N_, 0, nullptr, nullptr, nullptr, nullptr);
      k_init_maxden<<<(N_ * H_ + 255) / 256, 256, 0, stream>>>(MAXB, DEN);
      k_edge_score<<<eblk, 256, 0, stream>>>(FS, FD, src + (size_t)r * E_,
                                             dst + (size_t)r * E_,
                                             gat_attn + (size_t)r * NH_, SC, MAXB);
      k_edge_exp<<<eblk, 256, 0, stream>>>(dst + (size_t)r * E_, SC, MAXB, DEN);
      k_edge_agg<<<eblk, 256, 0, stream>>>(FS, src + (size_t)r * E_,
                                           dst + (size_t)r * E_, SC, DEN, HB, r);
    }
    // semantic attention weights (fused epilogue, result only via mean)
    gemm(HB, B_sem, nullptr, N_ * R_, 4, sem_b1, nullptr, sem_w2, WSUM);
    k_sem_beta<<<1, 1, 0, stream>>>(WSUM, BETA);
    k_inter_build<<<(N_ * NH_ + 255) / 256, 256, 0, stream>>>(
        HB, BETA, h_bias, BUF1 + (size_t)t * N_ * NH_);
  }

  // ---- phase B: temporal attention ----
  const int MT = T_ * N_;
  gemm(BUF1, B_proj, BUF2, MT, 3, proj_b, PE, nullptr, nullptr);   // HH = INTER@proj + b + pe
  gemm(BUF2, B_q, BUF1, MT, 0, nullptr, nullptr, nullptr, nullptr); // Q
  gemm(BUF2, B_k, BUF3, MT, 0, nullptr, nullptr, nullptr, nullptr); // K
  gemm(BUF2, B_v, VOUT, MT, 0, nullptr, nullptr, nullptr, nullptr); // V (in d_out)
  k_temp_attn<<<(N_ + 7) / 8, 256, 0, stream>>>(BUF1, BUF3, VOUT, BUF2); // CTX
  gemm(BUF2, B_fc, BUF1, MT, 2, fc_b, nullptr, nullptr, nullptr);  // HO = relu(CTX@fc + b)
  gemm(x,    B_res, BUF3, MT, 1, res_b, nullptr, nullptr, nullptr); // RES = x@res_w + b
  k_final_ln<<<(MT + 7) / 8, 256, 0, stream>>>(BUF1, BUF3, res_al, ln_gamma,
                                               ln_beta, (float*)d_out);
}
